// Net_22265110463047
// MI455X (gfx1250) — matrix operand order repaired and checked by reading, compiled, not run on hardware
//
#include <hip/hip_runtime.h>
#include <hip/hip_bf16.h>
#include <math.h>

typedef __bf16 bf16;
typedef __attribute__((ext_vector_type(16))) __bf16 v16bf;
typedef __attribute__((ext_vector_type(8)))  __bf16 v8bf;
typedef __attribute__((ext_vector_type(2)))  __bf16 v2bf;
typedef __attribute__((ext_vector_type(8)))  float  v8f;

#define NCLI   100          // n clients
#define NP     112          // padded to 7 tiles of 16
#define CHK    64           // K elements staged per LDS buffer
#define PITCH  72           // bf16 elements per LDS row (4/16B aligned, bank-skewed)
#define NSLAB  8            // gram atomic slabs (contention reduction)
#define GELEM  (NP*NP)

__global__ void zero_ws_kernel(float* g, int n) {
    int i = blockIdx.x * blockDim.x + threadIdx.x;
    if (i < n) g[i] = 0.f;
}

// ---------------------------------------------------------------------------
// Pass 1: G = X * X^T  (X = [n, d] client gradients; memory layout is [d, n])
// bf16 operands, f32 accumulation via v_wmma_f32_16x16x32_bf16.
// ---------------------------------------------------------------------------
__global__ __launch_bounds__(256) void gram_kernel(const float* __restrict__ x,
                                                   float* __restrict__ G8,
                                                   int d, int nchunks, int cpb) {
    __shared__ __align__(16) bf16 lds[2][NP * PITCH];

    const int tid  = threadIdx.x;
    const int lane = tid & 31;
    const int wvid = tid >> 5;      // wave id 0..7
    const int g    = lane >> 4;     // lane group (0: lanes 0-15, 1: lanes 16-31)
    const int mn   = lane & 15;     // M (for A) / N (for B) index within tile

    // ---- per-wave tile assignments: 28 upper-triangular pairs of 7x7 grid,
    //      computed arithmetically (no memory tables) and hoisted out of loop
    int tiS[4], tjS[4], aoff[4], boff[4];
    bool act[4];
    #pragma unroll
    for (int s = 0; s < 4; ++s) {
        int tt = wvid + s * 8;
        act[s] = (tt < 28);
        int t2 = act[s] ? tt : 0;
        int ti = (t2 >= 7) + (t2 >= 13) + (t2 >= 18) + (t2 >= 22) + (t2 >= 25) + (t2 >= 27);
        int tj = t2 - (ti * (15 - ti)) / 2 + ti;
        tiS[s] = ti; tjS[s] = tj;
        aoff[s] = (ti * 16 + mn) * PITCH + g * 8;   // A: elems {kk+g*8..+7, kk+16+g*8..+7}
        boff[s] = (tj * 16 + mn) * PITCH + g * 8;   // B: elems {kk+g*8..+7, kk+16+g*8..+7}
    }

    // ---- per-thread loader descriptors (chunk-invariant)
    //      item = (i4, k2): float4 along clients at rows k=2*k2 and k=2*k2+1
    //      28 float4-columns x 32 k-pairs = 896 items over 256 threads (<=4 each)
    int g4off[4], loff[4], lkk[4];
    bool sact[4], lval[4];
    #pragma unroll
    for (int it = 0; it < 4; ++it) {
        int l  = it * 256 + tid;
        int i4 = l % 28, k2 = l / 28;
        int k  = 2 * k2;
        sact[it]  = (l < 896);                 // must store (zero-pads rows 100..111)
        lval[it]  = sact[it] && (i4 < 25);     // has real data
        g4off[it] = k * (NCLI / 4) + i4;       // float4 index within chunk
        loff[it]  = (i4 * 4) * PITCH + k;      // bf16 element offset in LDS buffer
        lkk[it]   = k;
    }

    v8f acc[4];
    #pragma unroll
    for (int s = 0; s < 4; ++s)
        #pragma unroll
        for (int e = 0; e < 8; ++e) acc[s][e] = 0.f;

    int c0 = blockIdx.x * cpb;
    int c1 = c0 + cpb; if (c1 > nchunks) c1 = nchunks;

    float4 ra[4], rb[4];   // prefetch registers

    auto loadRegs = [&](int c) {
        const float4* px = (const float4*)(x + (size_t)c * CHK * NCLI);
        int krem = d - c * CHK;
        #pragma unroll
        for (int it = 0; it < 4; ++it) {
            float4 z; z.x = z.y = z.z = z.w = 0.f;
            ra[it] = z; rb[it] = z;
            if (lval[it]) {
                if (lkk[it]     < krem) ra[it] = px[g4off[it]];
                if (lkk[it] + 1 < krem) rb[it] = px[g4off[it] + NCLI / 4];
            }
        }
    };
    auto storeLds = [&](int p) {
        bf16* buf = &lds[p][0];
        #pragma unroll
        for (int it = 0; it < 4; ++it) {
            if (sact[it]) {
                float av[4] = { ra[it].x, ra[it].y, ra[it].z, ra[it].w };
                float bv[4] = { rb[it].x, rb[it].y, rb[it].z, rb[it].w };
                #pragma unroll
                for (int cc = 0; cc < 4; ++cc) {
                    v2bf pk;                          // pack (k, k+1) -> one dword
                    pk[0] = (bf16)av[cc];
                    pk[1] = (bf16)bv[cc];
                    *(v2bf*)&buf[loff[it] + cc * PITCH] = pk;
                }
            }
        }
    };
    auto compute = [&](int p) {
        const bf16* buf = &lds[p][0];
        #pragma unroll
        for (int s = 0; s < 4; ++s) {
            if (act[s]) {                           // uniform per wave -> EXEC all 1s
                #pragma unroll
                for (int kk = 0; kk < CHK; kk += 32) {
                    v8bf alo = *(const v8bf*)(buf + aoff[s] + kk);
                    v8bf ahi = *(const v8bf*)(buf + aoff[s] + kk + 16);
                    v8bf blo = *(const v8bf*)(buf + boff[s] + kk);
                    v8bf bhi = *(const v8bf*)(buf + boff[s] + kk + 16);
                    v16bf A, B;
                    #pragma unroll
                    for (int e = 0; e < 8; ++e) {
                        A[e] = alo[e]; A[e + 8] = ahi[e];
                        B[e] = blo[e]; B[e + 8] = bhi[e];
                    }
                    acc[s] = __builtin_amdgcn_wmma_f32_16x16x32_bf16(
                        false, A, false, B, (short)0, acc[s], false, false);
                }
            }
        }
    };

    if (c0 < c1) {
        loadRegs(c0);
        storeLds(0);
        __syncthreads();

        for (int c = c0; c < c1; ++c) {
            const int  p    = (c - c0) & 1;
            const bool more = (c + 1 < c1);
            if (more) loadRegs(c + 1);      // global b128 loads hide under WMMA
            compute(p);
            __syncthreads();
            if (more) {
                storeLds(p ^ 1);
                __syncthreads();
            }
        }
    }

    // ---- single atomic flush of f32 accumulators (8 slabs reduce contention) ----
    float* slab = G8 + (size_t)(blockIdx.x & (NSLAB - 1)) * GELEM;
    #pragma unroll
    for (int s = 0; s < 4; ++s) {
        if (act[s]) {
            #pragma unroll
            for (int rr = 0; rr < 8; ++rr) {
                int row = tiS[s] * 16 + rr + g * 8;   // C/D layout: M = vgpr + 8*lanegroup
                int col = tjS[s] * 16 + mn;           // N = lane % 16
                atomicAdd(&slab[row * NP + col], acc[s][rr]);
            }
        }
    }
}

__global__ void reduce_slabs_kernel(float* G8) {
    int i = blockIdx.x * blockDim.x + threadIdx.x;
    if (i < GELEM) {
        float s = 0.f;
        #pragma unroll
        for (int p = 0; p < NSLAB; ++p) s += G8[(size_t)p * GELEM + i];
        G8[i] = s;
    }
}

// ---------------------------------------------------------------------------
// Pass 2: FoolsGold weight computation on the 100x100 Gram matrix.
// G is stored upper-tile-triangular (tile(i) <= tile(j)); diagonal tiles full.
// ---------------------------------------------------------------------------
__global__ __launch_bounds__(128) void foolsgold_kernel(const float* __restrict__ G,
                                                        float* __restrict__ wOut) {
    __shared__ float sn[NCLI];
    __shared__ float smax[NCLI];
    __shared__ float sv[NCLI];
    __shared__ float red;
    int i = threadIdx.x;

    auto getG = [&](int a, int b) -> float {
        int ta = a >> 4, tb = b >> 4;
        return (tb < ta) ? G[b * NP + a] : G[a * NP + b];
    };

    if (i < NCLI) {
        float gii = getG(i, i);                            // ||g_i||^2 from diag(G)
        sn[i] = fmaxf(sqrtf(fmaxf(gii, 0.f)), 1e-12f);     // eps clamp as in F.normalize
    }
    __syncthreads();

    if (i < NCLI) {
        float mc = 0.f;                                    // diag of cs is 0 -> init 0
        for (int j = 0; j < NCLI; ++j)
            if (j != i) mc = fmaxf(mc, getG(i, j) / (sn[i] * sn[j]));
        smax[i] = mc;
    }
    __syncthreads();

    float wv = 0.f;
    if (i < NCLI) {
        float m2 = 0.f;                                    // includes diag 0
        for (int j = 0; j < NCLI; ++j) {
            if (j == i) continue;
            float c = getG(i, j) / (sn[i] * sn[j]);
            if (smax[i] < smax[j]) c *= smax[i] / smax[j]; // pardoning
            m2 = fmaxf(m2, c);
        }
        wv = 1.f - m2;
        wv = fminf(fmaxf(wv, 0.f), 1.f);
        sv[i] = wv;
    }
    __syncthreads();
    if (i == 0) {
        float mx = sv[0];
        for (int j = 1; j < NCLI; ++j) mx = fmaxf(mx, sv[j]);
        red = mx;
    }
    __syncthreads();

    float lv = 0.f;
    if (i < NCLI) {
        wv = wv / red;
        if (wv == 1.f) wv = 0.99f;
        lv = logf(wv / (1.f - wv)) + 0.5f;
        float infterm = isinf(lv) ? 1.f : 0.f;             // torch quirk: isinf + wv > 1
        if (infterm + lv > 1.f) lv = 1.f;
        if (lv < 0.f) lv = 0.f;
        sv[i] = lv;
    }
    __syncthreads();
    if (i == 0) {
        float s = 0.f;
        for (int j = 0; j < NCLI; ++j) s += sv[j];
        red = s;
    }
    __syncthreads();
    if (i < NCLI) wOut[i] = lv / red;                      // w = wv / sum(wv)
}

// ---------------------------------------------------------------------------
// Pass 3: out[k] = sum_i x[k, i] * w[i]   (k*100 floats is 16B-aligned -> float4)
// ---------------------------------------------------------------------------
__global__ __launch_bounds__(256) void wsum_kernel(const float* __restrict__ x,
                                                   const float* __restrict__ w,
                                                   float* __restrict__ out, int d) {
    __shared__ float sw[NCLI];
    if (threadIdx.x < NCLI) sw[threadIdx.x] = w[threadIdx.x];
    __syncthreads();
    int k = blockIdx.x * blockDim.x + threadIdx.x;
    if (k < d) {
        const float4* p = (const float4*)(x + (size_t)k * NCLI);
        float s = 0.f;
        #pragma unroll
        for (int j = 0; j < NCLI / 4; ++j) {
            float4 v = p[j];
            s += v.x * sw[4*j] + v.y * sw[4*j+1] + v.z * sw[4*j+2] + v.w * sw[4*j+3];
        }
        out[k] = s;
    }
}

extern "C" void kernel_launch(void* const* d_in, const int* in_sizes, int n_in,
                              void* d_out, int out_size, void* d_ws, size_t ws_size,
                              hipStream_t stream) {
    const float* x = (const float*)d_in[0];
    float* out = (float*)d_out;
    const int d = in_sizes[0] / NCLI;          // 1,000,000

    float* G8 = (float*)d_ws;                  // NSLAB * 112*112 f32 partial grams
    float* w  = G8 + (size_t)NSLAB * GELEM;    // 100 client weights

    // G slabs accumulate via atomics -> must be zeroed every call
    const int zn = NSLAB * GELEM;
    zero_ws_kernel<<<(zn + 255) / 256, 256, 0, stream>>>(G8, zn);

    int nchunks = (d + CHK - 1) / CHK;         // 15625
    int grid = 512; if (grid > nchunks) grid = nchunks;
    int cpb = (nchunks + grid - 1) / grid;
    gram_kernel<<<grid, 256, 0, stream>>>(x, G8, d, nchunks, cpb);

    reduce_slabs_kernel<<<(GELEM + 255) / 256, 256, 0, stream>>>(G8);
    foolsgold_kernel<<<1, 128, 0, stream>>>(G8, w);
    wsum_kernel<<<(d + 255) / 256, 256, 0, stream>>>(x, w, out, d);
}
